// EncoderProcesserDecoder_19834158973147
// MI455X (gfx1250) — compile-verified, matrix operand order checked by reading, not run on hardware
//
#include <hip/hip_runtime.h>
#include <hip/hip_bf16.h>
#include <stdint.h>

#define HDIM   128
#define NHEADS 4
#define DH     32
#define MPB    10
#define NN     20000
#define NE     120000

typedef __attribute__((ext_vector_type(16))) __bf16 v16bf;
typedef __attribute__((ext_vector_type(8)))  float  v8f;

struct alignas(16) U4 { unsigned int x, y, z, w; };
union Frag { v16bf v; U4 q[2]; };

__device__ __forceinline__ unsigned short f2bf(float f) {
    unsigned int u = __float_as_uint(f);
    u += 0x7FFFu + ((u >> 16) & 1u);          // round-to-nearest-even
    return (unsigned short)(u >> 16);
}
__device__ __forceinline__ float bf2f(unsigned short h) {
    return __uint_as_float(((unsigned int)h) << 16);
}

// ---------------------------------------------------------------------------
// Core WMMA GEMM:  C[M x 128] = act(A[M x K] @ W[K x 128] + b), optional LN.
// A, Wt in bf16 (Wt pre-transposed to 128 x K, row-major).
// Block = 256 threads = 8 waves; each wave owns a 16-col stripe; block does 32 rows.
// M % 32 == 0, K % 32 == 0 (holds for all uses here).
// ---------------------------------------------------------------------------
__global__ __launch_bounds__(256) void gemm128_bf16(
    const unsigned short* __restrict__ A, const unsigned short* __restrict__ Wt,
    const float* __restrict__ bias, int K,
    float* __restrict__ Cf, unsigned short* __restrict__ Cb,
    int relu, const float* __restrict__ ln_g, const float* __restrict__ ln_b)
{
    __shared__ float tile[32][132];
    const int wave = threadIdx.x >> 5;
    const int lane = threadIdx.x & 31;
    const int lh   = lane & 15;
    const int g    = lane >> 4;              // 0: K in {0-7,16-23}; 1: K in {8-15,24-31}
    const int m0   = blockIdx.x * 32 + lh;
    const int n    = wave * 16 + lh;         // global col 0..127

    const unsigned short* A0 = A  + (size_t)m0 * K + 8 * g;
    const unsigned short* A1 = A0 + (size_t)16 * K;
    const unsigned short* B0 = Wt + (size_t)n  * K + 8 * g;

    v8f acc0 = {0.f,0.f,0.f,0.f,0.f,0.f,0.f,0.f};
    v8f acc1 = {0.f,0.f,0.f,0.f,0.f,0.f,0.f,0.f};

    for (int k0 = 0; k0 < K; k0 += 32) {
        Frag a0, a1, b;
        a0.q[0] = *(const U4*)(A0 + k0);  a0.q[1] = *(const U4*)(A0 + k0 + 16);
        a1.q[0] = *(const U4*)(A1 + k0);  a1.q[1] = *(const U4*)(A1 + k0 + 16);
        b.q[0]  = *(const U4*)(B0 + k0);  b.q[1]  = *(const U4*)(B0 + k0 + 16);
        acc0 = __builtin_amdgcn_wmma_f32_16x16x32_bf16(false, a0.v, false, b.v,
                                                       (short)0, acc0, false, false);
        acc1 = __builtin_amdgcn_wmma_f32_16x16x32_bf16(false, a1.v, false, b.v,
                                                       (short)0, acc1, false, false);
    }

    const float bv = bias ? bias[n] : 0.0f;

    if (ln_g) {
        // stage fp32 tile in LDS, then fused LayerNorm over the 128 features
        #pragma unroll
        for (int r = 0; r < 8; ++r) {
            tile[r + 8 * g][n]      = acc0[r] + bv;
            tile[16 + r + 8 * g][n] = acc1[r] + bv;
        }
        __syncthreads();
        const int row = threadIdx.x >> 4;          // 0..15
        const int c0  = (threadIdx.x & 15) * 8;    // 0..120
        for (int rr = row; rr < 32; rr += 16) {
            float s = 0.f, s2 = 0.f;
            #pragma unroll
            for (int j = 0; j < 8; ++j) { float v = tile[rr][c0 + j]; s += v; s2 += v * v; }
            #pragma unroll
            for (int off = 1; off < 16; off <<= 1) {
                s  += __shfl_xor(s,  off, 32);
                s2 += __shfl_xor(s2, off, 32);
            }
            const float mu  = s * (1.0f / 128.0f);
            const float var = s2 * (1.0f / 128.0f) - mu * mu;
            const float inv = rsqrtf(var + 1e-5f);
            const size_t gm = (size_t)blockIdx.x * 32 + rr;
            #pragma unroll
            for (int j = 0; j < 8; ++j) {
                const int c = c0 + j;
                float v = (tile[rr][c] - mu) * inv * ln_g[c] + ln_b[c];
                if (Cf) Cf[gm * 128 + c] = v;
                if (Cb) Cb[gm * 128 + c] = f2bf(v);
            }
        }
    } else {
        #pragma unroll
        for (int r = 0; r < 8; ++r) {
            float v0 = acc0[r] + bv;
            float v1 = acc1[r] + bv;
            if (relu) { v0 = v0 > 0.f ? v0 : 0.f; v1 = v1 > 0.f ? v1 : 0.f; }
            const size_t i0 = ((size_t)blockIdx.x * 32 + r + 8 * g) * 128 + n;
            const size_t i1 = i0 + (size_t)16 * 128;
            if (Cf) { Cf[i0] = v0; Cf[i1] = v1; }
            if (Cb) { Cb[i0] = f2bf(v0); Cb[i1] = f2bf(v1); }
        }
    }
}

// fp32 weight (K x 128) -> bf16 transposed (128 x K)
__global__ void xpose_bf16(const float* __restrict__ W, unsigned short* __restrict__ Wt, int K) {
    int i = blockIdx.x * blockDim.x + threadIdx.x;
    if (i >= K * 128) return;
    int k = i / 128, n = i % 128;
    Wt[(size_t)n * K + k] = f2bf(W[(size_t)k * 128 + n]);
}

// tiny-K input layer: y = relu(x @ W + b), output bf16.  thread per (m, n)
__global__ void lin_small_relu(const float* __restrict__ X, const float* __restrict__ W,
                               const float* __restrict__ b, unsigned short* __restrict__ Y,
                               int M, int K) {
    int i = blockIdx.x * blockDim.x + threadIdx.x;
    if (i >= M * 128) return;
    int m = i / 128, n = i % 128;
    float s = b[n];
    for (int k = 0; k < K; ++k) s += X[(size_t)m * K + k] * W[(size_t)k * 128 + n];
    Y[i] = f2bf(s > 0.f ? s : 0.f);
}

// decoder head: out = h @ W(128x2) + b, fp32 out
__global__ void lin_final(const unsigned short* __restrict__ Hb, const float* __restrict__ W,
                          const float* __restrict__ b, float* __restrict__ out, int M) {
    int i = blockIdx.x * blockDim.x + threadIdx.x;
    if (i >= M * 2) return;
    int m = i >> 1, o = i & 1;
    float s = b[o];
    for (int k = 0; k < 128; ++k) s += bf2f(Hb[(size_t)m * 128 + k]) * W[k * 2 + o];
    out[i] = s;
}

__global__ void concat_col(const unsigned short* __restrict__ Eb, const unsigned short* __restrict__ Xb,
                           const int* __restrict__ snd, const int* __restrict__ rcv,
                           unsigned short* __restrict__ Col) {
    int i = blockIdx.x * blockDim.x + threadIdx.x;
    if (i >= NE * 128) return;
    int e = i >> 7, h = i & 127;
    unsigned short* c = Col + (size_t)e * 384;
    c[h]       = Eb[(size_t)e * 128 + h];
    c[128 + h] = Xb[(size_t)snd[e] * 128 + h];
    c[256 + h] = Xb[(size_t)rcv[e] * 128 + h];
}

__global__ void zero_f32(float* __restrict__ p, int n) {
    int i = blockIdx.x * blockDim.x + threadIdx.x;
    if (i < n) p[i] = 0.f;
}

__global__ void scatter_add(const float* __restrict__ Ef, const int* __restrict__ rcv,
                            float* __restrict__ Agg) {
    int i = blockIdx.x * blockDim.x + threadIdx.x;
    if (i >= NE * 128) return;
    int e = i >> 7, h = i & 127;
    atomicAdd(&Agg[(size_t)rcv[e] * 128 + h], Ef[i]);
}

__global__ void cvt_bf16(const float* __restrict__ x, unsigned short* __restrict__ y, int n) {
    int i = blockIdx.x * blockDim.x + threadIdx.x;
    if (i < n) y[i] = f2bf(x[i]);
}

// per-(node, head) attention over L cached memory slots (L <= 20)
__global__ void attn_core(const unsigned short* __restrict__ Qb, const unsigned short* __restrict__ Kc,
                          const unsigned short* __restrict__ Vc, unsigned short* __restrict__ Ob,
                          int L) {
    int i = blockIdx.x * blockDim.x + threadIdx.x;
    if (i >= NN * NHEADS) return;
    const int node = i >> 2, head = i & 3;
    const size_t base = (size_t)node * 128 + head * DH;
    float q[DH];
    #pragma unroll
    for (int d = 0; d < DH; ++d) q[d] = bf2f(Qb[base + d]);
    float sc[2 * MPB];
    float mx = -1e30f;
    const float scale = 0.17677669529663687f;   // 1/sqrt(32)
    for (int l = 0; l < L; ++l) {
        const unsigned short* kp = Kc + (size_t)l * NN * 128 + base;
        float s = 0.f;
        #pragma unroll
        for (int d = 0; d < DH; ++d) s += q[d] * bf2f(kp[d]);
        s *= scale;
        sc[l] = s;
        mx = fmaxf(mx, s);
    }
    float den = 0.f;
    for (int l = 0; l < L; ++l) { sc[l] = __expf(sc[l] - mx); den += sc[l]; }
    const float inv = 1.0f / den;
    float o[DH];
    #pragma unroll
    for (int d = 0; d < DH; ++d) o[d] = 0.f;
    for (int l = 0; l < L; ++l) {
        const unsigned short* vp = Vc + (size_t)l * NN * 128 + base;
        const float a = sc[l] * inv;
        #pragma unroll
        for (int d = 0; d < DH; ++d) o[d] += a * bf2f(vp[d]);
    }
    #pragma unroll
    for (int d = 0; d < DH; ++d) Ob[base + d] = f2bf(o[d]);
}

// ---------------------------------------------------------------------------
extern "C" void kernel_launch(void* const* d_in, const int* in_sizes, int n_in,
                              void* d_out, int out_size, void* d_ws, size_t ws_size,
                              hipStream_t stream) {
    (void)in_sizes; (void)out_size; (void)ws_size;
    if (n_in < 235) return;
    auto F = [&](int i) { return (const float*)d_in[i]; };

    // ---- input index map (insertion-order flatten of setup_inputs dict) ----
    const float* node_attr = F(0);
    const float* edge_attr = F(1);
    const int*   senders   = (const int*)d_in[2];
    const int*   receivers = senders + NE;
    // enc_node 3..12, enc_edge 13..22, dec 23..30, attn 31..34, blocks 35+20i

    // ---- workspace carve-up ----
    size_t off = 0;
    auto alloc = [&](size_t bytes) {
        char* p = (char*)d_ws + off;
        off += (bytes + 255) & ~(size_t)255;
        return p;
    };
    unsigned short* col    = (unsigned short*)alloc((size_t)NE * 384 * 2);
    unsigned short* t1     = (unsigned short*)alloc((size_t)NE * 128 * 2);
    unsigned short* t2     = (unsigned short*)alloc((size_t)NE * 128 * 2);
    unsigned short* e_bf   = (unsigned short*)alloc((size_t)NE * 128 * 2);
    float*          e_f    = (float*)        alloc((size_t)NE * 128 * 4);
    unsigned short* x_bf   = (unsigned short*)alloc((size_t)NN * 128 * 2);
    float*          agg_f  = (float*)        alloc((size_t)NN * 128 * 4);
    unsigned short* agg_bf = (unsigned short*)alloc((size_t)NN * 128 * 2);
    unsigned short* q_bf   = (unsigned short*)alloc((size_t)NN * 128 * 2);
    unsigned short* o_bf   = (unsigned short*)alloc((size_t)NN * 128 * 2);
    unsigned short* nd_bf  = (unsigned short*)alloc((size_t)NN * 128 * 2);
    unsigned short* Kc     = (unsigned short*)alloc((size_t)2 * MPB * NN * 128 * 2);
    unsigned short* Vc     = (unsigned short*)alloc((size_t)2 * MPB * NN * 128 * 2);
    unsigned short* wt     = (unsigned short*)alloc((size_t)2400000 * 2); // bf16 weight arena

    // ---- convert + transpose all GEMM weights to bf16 (N x K) ----
    size_t wcur = 0;
    auto allocW = [&](int K) { unsigned short* p = wt + wcur; wcur += (size_t)128 * K; return p; };
    auto xpose  = [&](const float* W, unsigned short* Wt, int K) {
        int n = K * 128;
        xpose_bf16<<<(n + 255) / 256, 256, 0, stream>>>(W, Wt, K);
    };
    unsigned short *encN_Wt[3], *encE_Wt[3], *dec_Wt[3], *WqT, *WkT, *WvT, *WoT;
    unsigned short *ebWt[MPB][4], *nbWt[MPB][4];
    for (int j = 0; j < 3; ++j) { encN_Wt[j] = allocW(128); xpose(F(5 + 2 * j),  encN_Wt[j], 128); }
    for (int j = 0; j < 3; ++j) { encE_Wt[j] = allocW(128); xpose(F(15 + 2 * j), encE_Wt[j], 128); }
    for (int j = 0; j < 3; ++j) { dec_Wt[j]  = allocW(128); xpose(F(23 + 2 * j), dec_Wt[j],  128); }
    WqT = allocW(128); xpose(F(31), WqT, 128);
    WkT = allocW(128); xpose(F(32), WkT, 128);
    WvT = allocW(128); xpose(F(33), WvT, 128);
    WoT = allocW(128); xpose(F(34), WoT, 128);
    for (int i = 0; i < MPB; ++i) {
        const int base = 35 + 20 * i;
        ebWt[i][0] = allocW(384); xpose(F(base + 0), ebWt[i][0], 384);
        for (int j = 1; j < 4; ++j) { ebWt[i][j] = allocW(128); xpose(F(base + 2 * j), ebWt[i][j], 128); }
        for (int j = 0; j < 4; ++j) { nbWt[i][j] = allocW(128); xpose(F(base + 10 + 2 * j), nbWt[i][j], 128); }
    }

    auto gemm = [&](const unsigned short* A, const unsigned short* Wt_, const float* bias,
                    int M, int K, float* Cf, unsigned short* Cb, int relu,
                    const float* g, const float* be) {
        gemm128_bf16<<<M / 32, 256, 0, stream>>>(A, Wt_, bias, K, Cf, Cb, relu, g, be);
    };

    // ---- node encoder: 11 -> 128 -> 128 -> 128 -> 128 + LN  (out: x_bf) ----
    lin_small_relu<<<(NN * 128 + 255) / 256, 256, 0, stream>>>(node_attr, F(3), F(4), t1, NN, 11);
    gemm(t1, encN_Wt[0], F(6),  NN, 128, nullptr, t2,   1, nullptr, nullptr);
    gemm(t2, encN_Wt[1], F(8),  NN, 128, nullptr, t1,   1, nullptr, nullptr);
    gemm(t1, encN_Wt[2], F(10), NN, 128, nullptr, x_bf, 0, F(11), F(12));

    // ---- edge encoder: 3 -> 128 x3 + LN  (out: e_bf) ----
    lin_small_relu<<<(NE * 128 + 255) / 256, 256, 0, stream>>>(edge_attr, F(13), F(14), t1, NE, 3);
    gemm(t1, encE_Wt[0], F(16), NE, 128, nullptr, t2,   1, nullptr, nullptr);
    gemm(t2, encE_Wt[1], F(18), NE, 128, nullptr, t1,   1, nullptr, nullptr);
    gemm(t1, encE_Wt[2], F(20), NE, 128, nullptr, e_bf, 0, F(21), F(22));

    // ---- message-passing blocks ----
    for (int i = 0; i < MPB; ++i) {
        const int base = 35 + 20 * i;
        // edge block: concat(e, x[snd], x[rcv]) -> MLP(384,128,128,128)+LN
        concat_col<<<(NE * 128 + 255) / 256, 256, 0, stream>>>(e_bf, x_bf, senders, receivers, col);
        gemm(col, ebWt[i][0], F(base + 1), NE, 384, nullptr, t1, 1, nullptr, nullptr);
        gemm(t1,  ebWt[i][1], F(base + 3), NE, 128, nullptr, t2, 1, nullptr, nullptr);
        gemm(t2,  ebWt[i][2], F(base + 5), NE, 128, nullptr, t1, 1, nullptr, nullptr);
        gemm(t1,  ebWt[i][3], F(base + 7), NE, 128, e_f, e_bf, 0, F(base + 8), F(base + 9));
        // agg = segment_sum(e, receivers)
        zero_f32<<<(NN * 128 + 255) / 256, 256, 0, stream>>>(agg_f, NN * 128);
        scatter_add<<<(NE * 128 + 255) / 256, 256, 0, stream>>>(e_f, receivers, agg_f);
        cvt_bf16<<<(NN * 128 + 255) / 256, 256, 0, stream>>>(agg_f, agg_bf, NN * 128);
        // cache K/V for the two new memory slots (2i: x, 2i+1: agg)
        unsigned short* K0 = Kc + (size_t)(2 * i)     * NN * 128;
        unsigned short* K1 = Kc + (size_t)(2 * i + 1) * NN * 128;
        unsigned short* V0 = Vc + (size_t)(2 * i)     * NN * 128;
        unsigned short* V1 = Vc + (size_t)(2 * i + 1) * NN * 128;
        gemm(x_bf,   WkT, nullptr, NN, 128, nullptr, K0, 0, nullptr, nullptr);
        gemm(agg_bf, WkT, nullptr, NN, 128, nullptr, K1, 0, nullptr, nullptr);
        gemm(x_bf,   WvT, nullptr, NN, 128, nullptr, V0, 0, nullptr, nullptr);
        gemm(agg_bf, WvT, nullptr, NN, 128, nullptr, V1, 0, nullptr, nullptr);
        // attention: Q = x@Wq; softmax over L slots; o@Wo
        gemm(x_bf, WqT, nullptr, NN, 128, nullptr, q_bf, 0, nullptr, nullptr);
        attn_core<<<(NN * NHEADS + 255) / 256, 256, 0, stream>>>(q_bf, Kc, Vc, o_bf, 2 * i + 2);
        gemm(o_bf, WoT, nullptr, NN, 128, nullptr, nd_bf, 0, nullptr, nullptr);
        // node block MLP + LN -> new x
        gemm(nd_bf, nbWt[i][0], F(base + 11), NN, 128, nullptr, t1,   1, nullptr, nullptr);
        gemm(t1,    nbWt[i][1], F(base + 13), NN, 128, nullptr, t2,   1, nullptr, nullptr);
        gemm(t2,    nbWt[i][2], F(base + 15), NN, 128, nullptr, t1,   1, nullptr, nullptr);
        gemm(t1,    nbWt[i][3], F(base + 17), NN, 128, nullptr, x_bf, 0, F(base + 18), F(base + 19));
    }

    // ---- decoder: 128 x3 relu, then 128 -> 2 ----
    gemm(x_bf, dec_Wt[0], F(24), NN, 128, nullptr, t1, 1, nullptr, nullptr);
    gemm(t1,   dec_Wt[1], F(26), NN, 128, nullptr, t2, 1, nullptr, nullptr);
    gemm(t2,   dec_Wt[2], F(28), NN, 128, nullptr, t1, 1, nullptr, nullptr);
    lin_final<<<(NN * 2 + 255) / 256, 256, 0, stream>>>(t1, F(29), F(30), (float*)d_out, NN);
}
